// VolatileFeatures_481036337348
// MI455X (gfx1250) — matrix operand order repaired and checked by reading, not compiled, not run on hardware
//
#include <hip/hip_runtime.h>

typedef __attribute__((ext_vector_type(16))) __bf16 v16bf;
typedef __attribute__((ext_vector_type(8)))  float  v8f;
typedef __attribute__((ext_vector_type(4)))  __bf16 v4bf;

#define LOOKBACK 512
#define D_MODEL  512
#define MODES    64
#define TM 128          // rows per workgroup
#define TN 128          // cols per workgroup
#define KSTEPS (LOOKBACK / 32)   // 16 K-steps of 32

// ---------------------------------------------------------------------------
// Async global->LDS copy (CDNA5 ASYNCcnt path), inline asm so it works on both
// ROCm 7.2 and amdgpu-toolchain regardless of builtin arity differences.
// ---------------------------------------------------------------------------
__device__ __forceinline__ void async_copy_b128(unsigned lds_byte_addr, const void* gaddr) {
    asm volatile("global_load_async_to_lds_b128 %0, %1, off"
                 :: "v"(lds_byte_addr), "v"(gaddr)
                 : "memory");
}

__device__ __forceinline__ void wait_async0() {
#if __has_builtin(__builtin_amdgcn_s_wait_asynccnt)
    __builtin_amdgcn_s_wait_asynccnt(0);
#else
    asm volatile("s_wait_asynccnt 0" ::: "memory");
#endif
}

// ---------------------------------------------------------------------------
// Kernel 1: build combined Fourier-projection matrix M (512x512) directly in
// WMMA B-fragment-packed bf16 layout:
//   Bp[kt][nt][lane][j],  kt=K-tile(32), nt=N-tile(16), lane 0..31, j 0..15
//   element = M[k][n],  n = nt*16 + (lane&15),  k = kt*32 + (lane>>4)*16 + j
// k order is contiguous (k = 16h + j), same as the A staging, so GEMM fragment
// loads are one contiguous 32B LDS read per lane.
// ---------------------------------------------------------------------------
__global__ __launch_bounds__(256) void build_basis_kernel(
    const float* __restrict__ W, void* __restrict__ wsv)
{
    __bf16* Bp = (__bf16*)wsv;
    const int idx  = blockIdx.x * 256 + threadIdx.x;   // 0 .. 262143
    const int j    = idx & 15;
    const int lane = (idx >> 4) & 31;
    const int nt   = (idx >> 9) & 31;
    const int kt   = idx >> 14;
    const int n    = nt * 16 + (lane & 15);            // output feature d
    const int k    = kt * 32 + (lane >> 4) * 16 + j;   // time index t

    const float w0 = 6.28318530717958647692f * (float)k / (float)LOOKBACK;
    float acc = 0.0f;
#pragma unroll 4
    for (int m = 0; m < MODES; ++m) {
        float s, c;
        __sincosf(w0 * (float)m, &s, &c);
        acc = fmaf(c, W[n * (2 * MODES) + m], acc);
        acc = fmaf(-s, W[n * (2 * MODES) + MODES + m], acc);
    }
    Bp[idx] = (__bf16)acc;
}

// ---------------------------------------------------------------------------
// Kernel 2: out = vol (Nx512 fp32) * M (512x512 bf16, packed) + bias
// 256 threads = 8 waves; wave grid 4x2; each wave owns a 32x64 output patch
// = 2x4 accumulator tiles of v_wmma_f32_16x16x32_bf16.
// B slab (512K x 128N = 128KB) loaded once via async-to-LDS; A double-buffered.
// ---------------------------------------------------------------------------
__global__ __launch_bounds__(256) void fourier_gemm_kernel(
    const float* __restrict__ A,       // vol, (nrows, 512) row-major fp32
    const void*  __restrict__ Bpv,     // packed bf16 basis (512KB)
    const float* __restrict__ bias,    // (512,)
    float* __restrict__ out,           // (nrows, 512)
    int nrows)
{
    const __bf16* Bp = (const __bf16*)Bpv;

    // LDS: B slab 16kt x 8nt x 32lane x 16 = 128KB ; A double buffer 2x8KB
    __shared__ __align__(64) __bf16 Bs[KSTEPS * 8 * 32 * 16];
    __shared__ __align__(64) __bf16 As[2][8 * 32 * 16];

    const int tid  = threadIdx.x;
    const int lane = tid & 31;
    const int wave = tid >> 5;        // 0..7
    const int wrow = wave >> 1;       // 0..3 -> row offset 32*wrow
    const int wcol = wave & 1;        // 0..1 -> col offset 64*wcol
    const int rowbase = blockIdx.x * TM;
    const int colbase = blockIdx.y * TN;

    // ---- issue async copy of the whole B slab for our 128 columns ----
    {
        const char* gB = (const char*)Bp;
#pragma unroll 4
        for (int c = 0; c < 32; ++c) {
            int idx  = tid + 256 * c;            // 0..8191 b128 chunks
            int half = idx & 1;                  // which 16B half of 32B row
            int bl   = (idx >> 1) & 31;          // lane slot
            int ntl  = (idx >> 6) & 7;           // local n-tile
            int kt   = idx >> 9;                 // k-tile
            unsigned ldsAddr = (unsigned)(size_t)&Bs[(((kt * 8 + ntl) * 32 + bl) * 16) + half * 8];
            size_t gOff = (size_t)(((((size_t)kt * 32 + (size_t)blockIdx.y * 8 + ntl) * 32 + bl) * 16) + half * 8) * 2;
            async_copy_b128(ldsAddr, gB + gOff);
        }
    }

    // ---- stage one 128x32 A tile (fp32 -> bf16) into WMMA-A packed layout ----
    // A-fragment layout (16x32 bf16): lane<16 holds row m, K {0..15};
    // lane>=16 holds row m, K {16..31}; 16 bf16 contiguous per lane.
    auto stageA = [&](int ks, int buf) {
#pragma unroll
        for (int c = 0; c < 4; ++c) {
            int i   = tid + 256 * c;             // 0..1023 float4 chunks
            int row = i >> 3;                    // 0..127
            int kk0 = (i & 7) * 4;               // 0..28, K within 32-tile
            const float4 v = *(const float4*)(A + (size_t)(rowbase + row) * LOOKBACK
                                                + (size_t)ks * 32 + kk0);
            int rt  = row >> 4;                  // which 16-row tile
            int m   = row & 15;
            int grp = kk0 >> 3;                  // 0..3
            int l   = (grp >> 1) ? (m + 16) : m;  // lane slot
            int j   = kk0 & 15;
            v4bf t = { (__bf16)v.x, (__bf16)v.y, (__bf16)v.z, (__bf16)v.w };
            *(v4bf*)&As[buf][(rt * 32 + l) * 16 + j] = t;
        }
    };

    stageA(0, 0);
    wait_async0();
    __syncthreads();

    v8f acc[2][4] = {};

    for (int ks = 0; ks < KSTEPS; ++ks) {
        if (ks + 1 < KSTEPS) stageA(ks + 1, (ks + 1) & 1);

        const __bf16* ab = &As[ks & 1][0];
        v16bf afrag[2];
#pragma unroll
        for (int i = 0; i < 2; ++i)
            afrag[i] = *(const v16bf*)&ab[((wrow * 2 + i) * 32 + lane) * 16];

        v16bf bfrag[4];
#pragma unroll
        for (int jn = 0; jn < 4; ++jn)
            bfrag[jn] = *(const v16bf*)&Bs[(((ks * 8) + wcol * 4 + jn) * 32 + lane) * 16];

#pragma unroll
        for (int i = 0; i < 2; ++i)
#pragma unroll
            for (int jn = 0; jn < 4; ++jn)
                acc[i][jn] = __builtin_amdgcn_wmma_f32_16x16x32_bf16(
                    /*neg_a=*/false, afrag[i], /*neg_b=*/false, bfrag[jn],
                    /*c_mod=*/(short)0, acc[i][jn],
                    /*reuse_a=*/false, /*reuse_b=*/false);

        __syncthreads();
    }

    // ---- epilogue: C layout (VGPR r: lanes0-15 -> M=r, lanes16-31 -> M=r+8)
#pragma unroll
    for (int i = 0; i < 2; ++i) {
        const int mrow = rowbase + wrow * 32 + i * 16 + ((lane >> 4) * 8);
#pragma unroll
        for (int jn = 0; jn < 4; ++jn) {
            const int col = colbase + wcol * 64 + jn * 16 + (lane & 15);
            const float bb = bias[col];
            float* op = out + (size_t)mrow * D_MODEL + col;
#pragma unroll
            for (int r = 0; r < 8; ++r)
                op[(size_t)r * D_MODEL] = acc[i][jn][r] + bb;
        }
    }
}

// ---------------------------------------------------------------------------
extern "C" void kernel_launch(void* const* d_in, const int* in_sizes, int n_in,
                              void* d_out, int out_size, void* d_ws, size_t ws_size,
                              hipStream_t stream)
{
    const float* vol  = (const float*)d_in[0];   // (64,2048,512) fp32
    const float* W    = (const float*)d_in[1];   // (512,128) fp32
    const float* bias = (const float*)d_in[2];   // (512,) fp32
    float* out = (float*)d_out;

    const int nrows = in_sizes[0] / LOOKBACK;    // 131072

    // ws layout: packed bf16 basis, 512*512*2 = 512 KB
    build_basis_kernel<<<dim3((LOOKBACK * D_MODEL) / 256), dim3(256), 0, stream>>>(W, d_ws);

    dim3 grid(nrows / TM, D_MODEL / TN);         // 1024 x 4
    fourier_gemm_kernel<<<grid, dim3(256), 0, stream>>>(vol, d_ws, bias, out, nrows);
}
